// Correlation_Transformer_70446053589137
// MI455X (gfx1250) — matrix operand order repaired and checked by reading, not compiled, not run on hardware
//
#include <hip/hip_runtime.h>

typedef __attribute__((ext_vector_type(16))) _Float16 v16h;
typedef __attribute__((ext_vector_type(8)))  _Float16 v8h;
typedef __attribute__((ext_vector_type(8)))  float    v8f;

#define Bn 8
#define Sn 32
#define Cn 3
#define Hn 256
#define Wn 256
#define Kn (Cn*Wn)   // 768

// -------------------------------------------------------------------------
// 1) reciprocal L2 norms of each (c,w)-column, per (n,h).  src: [N,C,H,W]
//    rnorm[n*H + h] = 1 / max(||src[n,:,h,:]||_2, 1e-12)
// -------------------------------------------------------------------------
__global__ void __launch_bounds__(256)
norm_kernel(const float* __restrict__ src, float* __restrict__ rnorm) {
  __shared__ float red[256];
  const int blk = blockIdx.x;          // n*Hn + h
  const int n   = blk >> 8;
  const int h   = blk & 255;
  const int tid = threadIdx.x;
  float s = 0.f;
  const float* base = src + (((size_t)n * Cn) * Hn + h) * Wn + tid;
  #pragma unroll
  for (int c = 0; c < Cn; ++c) {
    float x = base[(size_t)c * Hn * Wn];
    s += x * x;
  }
  red[tid] = s;
  __syncthreads();
  for (int st = 128; st > 0; st >>= 1) {
    if (tid < st) red[tid] += red[tid + st];
    __syncthreads();
  }
  if (tid == 0) rnorm[blk] = 1.0f / fmaxf(sqrtf(red[0]), 1e-12f);
}

// -------------------------------------------------------------------------
// 2) batched GEMM G = A^T T (per (b,s): M=N=256, K=768) via WMMA f16,
//    scaled by inverse norms, reduced to per-region diagonal partial sums.
//    grid: (16 regions of 64x64, B*S).  block: 256 threads = 8 waves.
//    part[(bs*16+reg)*128 + dl] = sum over the region of diagonal dl-63.
// -------------------------------------------------------------------------
__global__ void __launch_bounds__(256)
corr_gemm_kernel(const float* __restrict__ input, const float* __restrict__ target,
                 const float* __restrict__ rni, const float* __restrict__ rnt,
                 float* __restrict__ part) {
  __shared__ __align__(16) _Float16 As[64][32];   // [m_local][k]   4 KB
  __shared__ __align__(16) _Float16 Ts[64][32];   // [n_local][k]   4 KB
  __shared__ float Gs[64][64];                    // scaled tile   16 KB
  __shared__ float rniS[64];
  __shared__ float rntS[64];

  const int bs  = blockIdx.y;          // b*Sn + s
  const int b   = bs >> 5;
  const int reg = blockIdx.x;          // 0..15
  const int mb  = (reg & 3) * 64;      // M (h1 of input) base
  const int nb  = (reg >> 2) * 64;     // N (h2 of target) base
  const int tid  = threadIdx.x;
  const int lane = tid & 31;
  const int wv   = tid >> 5;
  const int mt   = wv & 3;             // wave's 16-row M tile inside region
  const int ntb  = (wv >> 2) << 1;     // wave's first 16-col N tile (covers 2)

  if (tid < 64) {
    rniS[tid] = rni[bs * Hn + mb + tid];
    rntS[tid] = rnt[b  * Hn + nb + tid];
  }

  v8f acc0 = {};
  v8f acc1 = {};

  const int sm = tid >> 2;             // staging row   0..63
  const int sk = (tid & 3) << 3;       // staging k off 0,8,16,24

  for (int kb = 0; kb < Kn; kb += 32) {
    const int c     = kb >> 8;         // 32 | 256 -> slice never straddles c
    const int wbase = kb & 255;
    __syncthreads();                   // protect previous iteration's reads
    {
      const float* ip = input  + ((((size_t)bs * Cn + c) * Hn + (mb + sm)) * Wn + wbase + sk);
      const float* tp = target + ((((size_t)b  * Cn + c) * Hn + (nb + sm)) * Wn + wbase + sk);
      float4 f0 = *(const float4*)(ip);
      float4 f1 = *(const float4*)(ip + 4);
      float4 t0 = *(const float4*)(tp);
      float4 t1 = *(const float4*)(tp + 4);
      As[sm][sk + 0] = (_Float16)f0.x; As[sm][sk + 1] = (_Float16)f0.y;
      As[sm][sk + 2] = (_Float16)f0.z; As[sm][sk + 3] = (_Float16)f0.w;
      As[sm][sk + 4] = (_Float16)f1.x; As[sm][sk + 5] = (_Float16)f1.y;
      As[sm][sk + 6] = (_Float16)f1.z; As[sm][sk + 7] = (_Float16)f1.w;
      Ts[sm][sk + 0] = (_Float16)t0.x; Ts[sm][sk + 1] = (_Float16)t0.y;
      Ts[sm][sk + 2] = (_Float16)t0.z; Ts[sm][sk + 3] = (_Float16)t0.w;
      Ts[sm][sk + 4] = (_Float16)t1.x; Ts[sm][sk + 5] = (_Float16)t1.y;
      Ts[sm][sk + 6] = (_Float16)t1.z; Ts[sm][sk + 7] = (_Float16)t1.w;
    }
    __syncthreads();

    // A fragment: 16x32 (MxK). lane<16 -> row=lane, K 0..15;
    //             lane>=16 -> row=lane-16, K 16..31
    const int rowA = mt * 16 + (lane & 15);
    const int hs   = (lane >> 4) << 4;           // 0 or 16
    v8h alo = *(const v8h*)&As[rowA][hs];
    v8h ahi = *(const v8h*)&As[rowA][hs + 8];
    v16h a  = __builtin_shufflevector(alo, ahi, 0,1,2,3,4,5,6,7,8,9,10,11,12,13,14,15);

    // B fragment: 32x16 (KxN), B[k][n] = Ts[n][k]. lane<16 -> col=lane, K 0..15;
    //             lane>=16 -> col=lane-16, K 16..31
    const int kbB  = (lane >> 4) << 4;           // 0 or 16
    const int col0 = ntb * 16 + (lane & 15);
    v8h b0lo = *(const v8h*)&Ts[col0][kbB];
    v8h b0hi = *(const v8h*)&Ts[col0][kbB + 8];
    v16h b0  = __builtin_shufflevector(b0lo, b0hi, 0,1,2,3,4,5,6,7,8,9,10,11,12,13,14,15);
    acc0 = __builtin_amdgcn_wmma_f32_16x16x32_f16(false, a, false, b0, (short)0, acc0, false, false);

    const int col1 = col0 + 16;
    v8h b1lo = *(const v8h*)&Ts[col1][kbB];
    v8h b1hi = *(const v8h*)&Ts[col1][kbB + 8];
    v16h b1  = __builtin_shufflevector(b1lo, b1hi, 0,1,2,3,4,5,6,7,8,9,10,11,12,13,14,15);
    acc1 = __builtin_amdgcn_wmma_f32_16x16x32_f16(false, a, false, b1, (short)0, acc1, false, false);
  }
  __syncthreads();

  // C layout: VGPR r -> M = r (lanes 0-15) / r+8 (lanes 16-31); N = lane&15.
  {
    const int m_b  = mt * 16 + ((lane >> 4) << 3);
    const int n_l0 = ntb * 16 + (lane & 15);
    const int n_l1 = n_l0 + 16;
    #pragma unroll
    for (int r = 0; r < 8; ++r) {
      const int mm = m_b + r;
      Gs[mm][n_l0] = acc0[r] * rniS[mm] * rntS[n_l0];
      Gs[mm][n_l1] = acc1[r] * rniS[mm] * rntS[n_l1];
    }
  }
  __syncthreads();

  // deterministic per-diagonal sums inside this 64x64 region
  if (tid < 128) {
    float s = 0.f;
    if (tid < 127) {
      const int off = tid - 63;                  // local diagonal offset
      const int m0  = off < 0 ? -off : 0;
      const int m1  = off > 0 ? 64 - off : 64;
      for (int m = m0; m < m1; ++m) s += Gs[m][m + off];
    }
    part[((size_t)bs * 16 + reg) * 128 + tid] = s;
  }
}

// -------------------------------------------------------------------------
// 3) sum region partials per global diagonal d (fixed order -> deterministic),
//    divide by overlap (H - d), argmax with JAX tie-break (largest d).
// -------------------------------------------------------------------------
__global__ void __launch_bounds__(256)
reduce_argmax_kernel(const float* __restrict__ part, int* __restrict__ shiftbuf) {
  __shared__ float sv[256];
  __shared__ int   sd[256];
  const int bs = blockIdx.x;
  const int d  = threadIdx.x;
  float sum = 0.f;
  #pragma unroll
  for (int reg = 0; reg < 16; ++reg) {
    const int mb = (reg & 3) * 64;
    const int nb = (reg >> 2) * 64;
    const int dl = d - (nb - mb) + 63;
    if (dl >= 0 && dl < 127) sum += part[((size_t)bs * 16 + reg) * 128 + dl];
  }
  sv[d] = sum / (float)(Hn - d);   // mean over overlap size (H - d) == (p+1)
  sd[d] = d;
  __syncthreads();
  for (int st = 128; st > 0; st >>= 1) {
    if (d < st) {
      if (sv[d + st] > sv[d] || (sv[d + st] == sv[d] && sd[d + st] > sd[d])) {
        sv[d] = sv[d + st];
        sd[d] = sd[d + st];
      }
    }
    __syncthreads();
  }
  if (d == 0) shiftbuf[bs] = sd[0];   // shift = best diagonal d
}

// -------------------------------------------------------------------------
// 4) out[b,s,c,r,:] = input[b,s,c,r+shift,:] if r+shift < H else 0  (float4)
// -------------------------------------------------------------------------
__global__ void __launch_bounds__(256)
shift_copy_kernel(const float4* __restrict__ in4, const int* __restrict__ shiftbuf,
                  float4* __restrict__ out4) {
  const size_t t   = (size_t)blockIdx.x * blockDim.x + threadIdx.x;
  const int    w4  = (int)(t & 63);
  const int    r   = (int)((t >> 6) & 255);
  const size_t bsc = t >> 14;                 // bs*Cn + c
  const int    bs  = (int)(bsc / Cn);
  const int    sr  = r + shiftbuf[bs];
  float4 v = make_float4(0.f, 0.f, 0.f, 0.f);
  if (sr < Hn) v = in4[(bsc << 14) | ((size_t)sr << 6) | (size_t)w4];
  out4[t] = v;
}

// -------------------------------------------------------------------------
extern "C" void kernel_launch(void* const* d_in, const int* in_sizes, int n_in,
                              void* d_out, int out_size, void* d_ws, size_t ws_size,
                              hipStream_t stream) {
  const float* input  = (const float*)d_in[0];   // [B,S,C,H,W]
  const float* target = (const float*)d_in[1];   // [B,C,H,W]
  float* ws    = (float*)d_ws;
  float* rnt   = ws;                             // B*H            =   2048 f
  float* rni   = ws + 2048;                      // B*S*H          =  65536 f
  float* part  = ws + 2048 + 65536;              // B*S*16*128     = 524288 f
  int*   shift = (int*)(ws + 2048 + 65536 + 524288);  // B*S       =    256 i

  norm_kernel<<<Bn * Hn,      256, 0, stream>>>(target, rnt);
  norm_kernel<<<Bn * Sn * Hn, 256, 0, stream>>>(input,  rni);
  corr_gemm_kernel<<<dim3(16, Bn * Sn), 256, 0, stream>>>(input, target, rni, rnt, part);
  reduce_argmax_kernel<<<Bn * Sn, 256, 0, stream>>>(part, shift);
  const int total4 = Bn * Sn * Cn * Hn * (Wn / 4);   // 12,582,912
  shift_copy_kernel<<<total4 / 256, 256, 0, stream>>>((const float4*)input, shift,
                                                      (float4*)d_out);
}
